// RosaAttention_51943334478531
// MI455X (gfx1250) — matrix operand order repaired and checked by reading, not compiled, not run on hardware
//
#include <hip/hip_runtime.h>
#include <hip/hip_bf16.h>

// ---------------------------------------------------------------------------
// RosaAttention fused pipeline for MI455X (gfx1250, wave32, WMMA 16x16x32)
//   B=2 S=2048 D=1024 H=16 KV=4 QKB=16 VB=32 N_REP=4 TAU=1
// Key facts exploited:
//   * score = expected #matching bits in [0, 16]  -> softmax with STATIC max
//     (no online-max, no rescale, row-sum deferred to epilogue)
//   * 4 query heads per kv-group share pk/pv      -> 16 WMMAs per K/V load
// ---------------------------------------------------------------------------

typedef _Float16 f16;
typedef __attribute__((ext_vector_type(16))) _Float16 v16h;
typedef __attribute__((ext_vector_type(16))) __bf16   v16bf;
typedef __attribute__((ext_vector_type(8)))  float    v8f;

#define Bq   2
#define Sq   2048
#define Dq   1024
#define Hq   16
#define KVq  4
#define QKB  16
#define VBq  32

__device__ __forceinline__ float redsum16(float v) {
  v += __shfl_xor(v, 1, 32);
  v += __shfl_xor(v, 2, 32);
  v += __shfl_xor(v, 4, 32);
  v += __shfl_xor(v, 8, 32);
  return v;
}
__device__ __forceinline__ float fast_sigmoid(float x) {
  return __builtin_amdgcn_rcpf(1.0f + __expf(-x));   // v_exp_f32 + v_rcp_f32
}

// ---------------------------------------------------------------------------
// Kernel 1: fused QKV projection  (f16 WMMA, fp32 acc) + sigmoid + row sums.
// N tiles: 0..15 -> Q heads, 16..19 -> K kv-heads, 20..27 -> V kv-head halves.
// One wave per 16x16 output tile; K=1024 in 32 steps of 32.
// ---------------------------------------------------------------------------
__global__ void __launch_bounds__(128)
qkv_proj_kernel(const float* __restrict__ hidden,
                const float* __restrict__ Wq,
                const float* __restrict__ Wk,
                const float* __restrict__ Wv,
                f16* __restrict__ pq, f16* __restrict__ pkT, f16* __restrict__ pv,
                float* __restrict__ sq, float* __restrict__ sk)
{
  const int lane = threadIdx.x & 31;
  const int wid  = threadIdx.x >> 5;
  const int w    = blockIdx.x * 4 + wid;          // 1792 blocks * 4 waves = 7168
  const int mt   = w / 28, nt = w % 28;
  const int g    = lane >> 4, c = lane & 15;
  const int row0 = mt * 16;                        // global row in [0, B*S)

  const float* W; int ldw, col0;
  if (nt < 16)      { W = Wq; ldw = Hq  * QKB; col0 = nt * 16;        }
  else if (nt < 20) { W = Wk; ldw = KVq * QKB; col0 = (nt - 16) * 16; }
  else              { W = Wv; ldw = KVq * VBq; col0 = (nt - 20) * 16; }

  v8f acc = {};
  for (int kb = 0; kb < Dq; kb += 32) {
    v16h a, bfrag;
    const float* arow = hidden + (size_t)(row0 + c) * Dq + kb;
#pragma unroll
    for (int h = 0; h < 8; ++h) {
      a[h]     = (f16)arow[8 * g + h];             // K = kb + 8g + h
      a[h + 8] = (f16)arow[16 + 8 * g + h];        // K = kb + 16 + 8g + h
    }
    const float* brow = W + (size_t)kb * ldw + col0 + c;  // lane = N col, elems = K (table order)
#pragma unroll
    for (int n = 0; n < 16; ++n) bfrag[n] = (f16)brow[(size_t)(8 * g + (n & 7) + ((n >> 3) << 4)) * ldw];
    acc = __builtin_amdgcn_wmma_f32_16x16x32_f16(false, a, false, bfrag,
                                                 (short)0, acc, false, false);
  }

  float p[8];
#pragma unroll
  for (int r = 0; r < 8; ++r) p[r] = fast_sigmoid(acc[r]);

  if (nt < 16) {                                   // Q: pq[row][h][bit], sq[row][h]
    const int h = nt;
#pragma unroll
    for (int r = 0; r < 8; ++r) {
      const int row = row0 + r + 8 * g;
      pq[(size_t)row * (Hq * QKB) + h * QKB + c] = (f16)p[r];
      const float s = redsum16(p[r]);
      if (c == 0) sq[row * Hq + h] = s;
    }
  } else if (nt < 20) {                            // K: pkT[b][kv][bit][s], sk[row][kv]
    const int kv = nt - 16;
#pragma unroll
    for (int r = 0; r < 8; ++r) {
      const int row = row0 + r + 8 * g;
      const int b = row >> 11, s_ = row & (Sq - 1);
      pkT[((size_t)((b * KVq + kv) * QKB + c)) * Sq + s_] = (f16)p[r];
      const float s = redsum16(p[r]);
      if (c == 0) sk[row * KVq + kv] = s;
    }
  } else {                                         // V: pv[b][kv][s][vb]
    const int kv  = (nt - 20) >> 1;
    const int vb0 = ((nt - 20) & 1) * 16;
#pragma unroll
    for (int r = 0; r < 8; ++r) {
      const int row = row0 + r + 8 * g;
      const int b = row >> 11, s_ = row & (Sq - 1);
      pv[((size_t)((b * KVq + kv) * Sq + s_)) * VBq + vb0 + c] = (f16)p[r];
    }
  }
}

// ---------------------------------------------------------------------------
// Kernel 2: fold value-embedding affine into output weights.
//   bias[n] = sum_k e0[k]*Wo[k][n] ;  W2[k][n] = (e1[k]-e0[k])*Wo[k][n]  (bf16)
// ---------------------------------------------------------------------------
__global__ void prep_out_kernel(const float* __restrict__ Wo,
                                const float* __restrict__ e0,
                                const float* __restrict__ e1,
                                __bf16* __restrict__ W2, float* __restrict__ bias)
{
  const int n = blockIdx.x * blockDim.x + threadIdx.x;
  if (n >= Dq) return;
  float b = 0.f;
  for (int k = 0; k < Hq * VBq; ++k) {
    const float w = Wo[(size_t)k * Dq + n];
    b += e0[k] * w;
    W2[(size_t)k * Dq + n] = (__bf16)((e1[k] - e0[k]) * w);
  }
  bias[n] = b;
}

// ---------------------------------------------------------------------------
// Kernel 3: causal attention, one wave per (b, kv-group, 16-row q-tile),
// processing all 4 query heads of the kv-group to amortize K/V loads.
//   score in [0,16]  ->  p = exp(score - 16), static max, no rescaling.
//   P re-laid-out through LDS (4 tiles, one s_wait_dscnt) for the PV WMMA.
// ---------------------------------------------------------------------------
__global__ void __launch_bounds__(32)
rosa_attn_kernel(const f16* __restrict__ pq, const f16* __restrict__ pkT,
                 const f16* __restrict__ pv, const float* __restrict__ sq,
                 const float* __restrict__ sk, __bf16* __restrict__ vh)
{
  __shared__ _Float16 plds[4][16 * 32];            // P tiles: [head][q row][kv]
  const int lane = threadIdx.x & 31;
  const int g = lane >> 4, c = lane & 15;
  const int blk = blockIdx.x;                      // B*KV*128 = 1024
  const int t  = blk & 127;
  const int kv = (blk >> 7) & (KVq - 1);
  const int b  = blk >> 9;
  const int q0 = t * 16;

  // Per-head Q A-fragments (halves 8..15 are K-padding) and row sums.
  v16h  aq[4];
  float sqv[4][8];
  float lp[4][8];                                  // per-lane partial row sums
  v8f   O[4][2] = {};
#pragma unroll
  for (int hh = 0; hh < 4; ++hh) {
    const int h = kv * 4 + hh;
    const f16* qrow = pq + ((size_t)(b * Sq + q0 + c)) * (Hq * QKB) + h * QKB + 8 * g;
#pragma unroll
    for (int i = 0; i < 8; ++i) { aq[hh][i] = qrow[i]; aq[hh][i + 8] = (f16)0.f; }
#pragma unroll
    for (int r = 0; r < 8; ++r) {
      sqv[hh][r] = sq[(b * Sq + q0 + r + 8 * g) * Hq + h];
      lp[hh][r]  = 0.f;
    }
  }

  const int nblk = (t + 2) >> 1;                   // ceil((t+1)/2) blocks of 32 kv
  const f16* pkTh = pkT + (size_t)((b * KVq + kv) * QKB) * Sq;
  const f16* pvh  = pv  + (size_t)(b * KVq + kv) * Sq * VBq;

  for (int kb = 0; kb < nblk; ++kb) {
    const int kbase = kb * 32;

    // K B-fragments: lane = kv col, elems 0..7 = bits 8g..8g+7, elems 8..15 = zero-pad.
    v16h bk0 = {}, bk1 = {};
    {
      const f16* kr = pkTh + (size_t)(8 * g) * Sq + kbase + c;
#pragma unroll
      for (int n = 0; n < 8; ++n) { bk0[n] = kr[(size_t)n * Sq]; bk1[n] = kr[(size_t)n * Sq + 16]; }
    }
    // V B-fragments: lane = vb column, elems = kv rows (table order).
    v16h bv0, bv1;
    const f16* vr = pvh + (size_t)kbase * VBq + c;
#pragma unroll
    for (int n = 0; n < 16; ++n) { const int kk = 8 * g + (n & 7) + ((n >> 3) << 4); bv0[n] = vr[kk * VBq]; bv1[n] = vr[kk * VBq + 16]; }

    const int col0 = kbase + c, col1 = col0 + 16;
    const float sk0 = sk[(b * Sq + col0) * KVq + kv];
    const float sk1 = sk[(b * Sq + col1) * KVq + kv];

#pragma unroll
    for (int hh = 0; hh < 4; ++hh) {
      v8f c0 = {}, c1 = {};
      c0 = __builtin_amdgcn_wmma_f32_16x16x32_f16(false, aq[hh], false, bk0,
                                                  (short)0, c0, false, false);
      c1 = __builtin_amdgcn_wmma_f32_16x16x32_f16(false, aq[hh], false, bk1,
                                                  (short)0, c1, false, false);
#pragma unroll
      for (int r = 0; r < 8; ++r) {
        const int row = q0 + r + 8 * g;
        // score - 16 = -sq_i - sk_j + 2*dot  (score bounded by 16: static max)
        const float s0 = (col0 <= row) ? (2.f * c0[r] - sqv[hh][r] - sk0) : -3.0e38f;
        const float s1 = (col1 <= row) ? (2.f * c1[r] - sqv[hh][r] - sk1) : -3.0e38f;
        const float p0 = __expf(s0);               // masked -> exp(-huge) = 0
        const float p1 = __expf(s1);
        lp[hh][r] += p0 + p1;
        plds[hh][(r + 8 * g) * 32 + c]      = (f16)p0;
        plds[hh][(r + 8 * g) * 32 + 16 + c] = (f16)p1;
      }
    }
    // One wait covers all 4 cross-lane P transposes.
    asm volatile("s_wait_dscnt 0" ::: "memory");
    __builtin_amdgcn_wave_barrier();

#pragma unroll
    for (int hh = 0; hh < 4; ++hh) {
      v16h ap;                                     // lane = q row, halves = kv
#pragma unroll
      for (int i = 0; i < 8; ++i) {
        ap[i]     = plds[hh][c * 32 + 8 * g + i];
        ap[i + 8] = plds[hh][c * 32 + 16 + 8 * g + i];
      }
      O[hh][0] = __builtin_amdgcn_wmma_f32_16x16x32_f16(false, ap, false, bv0,
                                                        (short)0, O[hh][0], false, false);
      O[hh][1] = __builtin_amdgcn_wmma_f32_16x16x32_f16(false, ap, false, bv1,
                                                        (short)0, O[hh][1], false, false);
    }
    __builtin_amdgcn_wave_barrier();               // reads precede next-iter writes
  }

  // Epilogue: finish deferred row sums, normalize, emit vh in [0,1]
  // (e0/e1 affine folded into W2 + bias downstream).
#pragma unroll
  for (int hh = 0; hh < 4; ++hh) {
    const int h = kv * 4 + hh;
#pragma unroll
    for (int r = 0; r < 8; ++r) {
      const int row = q0 + r + 8 * g;
      const float l   = redsum16(lp[hh][r]);       // > 0: diagonal always present
      const float inv = __builtin_amdgcn_rcpf(l);
      __bf16* o = vh + ((size_t)(b * Sq + row)) * (Hq * VBq) + h * VBq;
      o[c]      = (__bf16)(O[hh][0][r] * inv);
      o[16 + c] = (__bf16)(O[hh][1][r] * inv);
    }
  }
}

// ---------------------------------------------------------------------------
// Kernel 4: output projection  out = bias + vh @ W2  (bf16 WMMA, K=512).
// ---------------------------------------------------------------------------
__global__ void __launch_bounds__(128)
out_proj_kernel(const __bf16* __restrict__ vh, const __bf16* __restrict__ W2,
                const float* __restrict__ bias, float* __restrict__ out)
{
  const int lane = threadIdx.x & 31;
  const int wid  = threadIdx.x >> 5;
  const int w    = blockIdx.x * 4 + wid;           // 4096 blocks * 4 = 16384 tiles
  const int mt   = w >> 6, nt = w & 63;
  const int g    = lane >> 4, c = lane & 15;
  const int row0 = mt * 16, n0 = nt * 16;

  v8f acc = {};
  for (int kb = 0; kb < Hq * VBq; kb += 32) {
    v16bf a, bf;
    const __bf16* ar = vh + (size_t)(row0 + c) * (Hq * VBq) + kb;
#pragma unroll
    for (int i = 0; i < 8; ++i) {
      a[i]     = ar[8 * g + i];
      a[i + 8] = ar[16 + 8 * g + i];
    }
    const __bf16* br = W2 + (size_t)kb * Dq + n0 + c;
#pragma unroll
    for (int n = 0; n < 16; ++n) bf[n] = br[(size_t)(8 * g + (n & 7) + ((n >> 3) << 4)) * Dq];
    acc = __builtin_amdgcn_wmma_f32_16x16x32_bf16(false, a, false, bf,
                                                  (short)0, acc, false, false);
  }
  const float bn = bias[n0 + c];
#pragma unroll
  for (int r = 0; r < 8; ++r)
    out[(size_t)(row0 + r + 8 * g) * Dq + n0 + c] = acc[r] + bn;
}

// ---------------------------------------------------------------------------
extern "C" void kernel_launch(void* const* d_in, const int* in_sizes, int n_in,
                              void* d_out, int out_size, void* d_ws, size_t ws_size,
                              hipStream_t stream) {
  const float* hidden = (const float*)d_in[0];
  const float* Wq = (const float*)d_in[1];
  const float* Wk = (const float*)d_in[2];
  const float* Wv = (const float*)d_in[3];
  const float* Wo = (const float*)d_in[4];
  const float* e0 = (const float*)d_in[5];
  const float* e1 = (const float*)d_in[6];
  float* out = (float*)d_out;

  // Workspace carve-out (~9.3 MB total).
  char*  ws  = (char*)d_ws;
  size_t off = 0;
  auto alloc = [&](size_t bytes) -> void* {
    void* p = ws + off;
    off += (bytes + 255) & ~(size_t)255;
    return p;
  };
  f16*    pq   = (f16*)   alloc((size_t)Bq * Sq * Hq  * QKB * sizeof(f16));
  f16*    pkT  = (f16*)   alloc((size_t)Bq * KVq * QKB * Sq * sizeof(f16));
  f16*    pv   = (f16*)   alloc((size_t)Bq * KVq * Sq * VBq * sizeof(f16));
  float*  sq   = (float*) alloc((size_t)Bq * Sq * Hq  * sizeof(float));
  float*  sk   = (float*) alloc((size_t)Bq * Sq * KVq * sizeof(float));
  __bf16* vh   = (__bf16*)alloc((size_t)Bq * Sq * Hq * VBq * sizeof(__bf16));
  __bf16* W2   = (__bf16*)alloc((size_t)Hq * VBq * Dq * sizeof(__bf16));
  float*  bias = (float*) alloc((size_t)Dq * sizeof(float));
  (void)ws_size; (void)in_sizes; (void)n_in; (void)out_size;

  // 1) QKV projections: 256 M-tiles x 28 N-tiles = 7168 waves, 4 waves/block.
  qkv_proj_kernel<<<1792, 128, 0, stream>>>(hidden, Wq, Wk, Wv, pq, pkT, pv, sq, sk);
  // 2) Fold e0/e1 into output weights + bias.
  prep_out_kernel<<<4, 256, 0, stream>>>(Wo, e0, e1, W2, bias);
  // 3) Attention: B*KV*(S/16) = 1024 single-wave workgroups, 4 heads each.
  rosa_attn_kernel<<<1024, 32, 0, stream>>>(pq, pkT, pv, sq, sk, vh);
  // 4) Output projection: 256 x 64 tiles, 4 waves/block.
  out_proj_kernel<<<4096, 128, 0, stream>>>(vh, W2, bias, out);
}
